// EGNNSteinCV_9414568313007
// MI455X (gfx1250) — matrix operand order repaired and checked by reading, not compiled, not run on hardware
//
// EGNN forward, fused per-batch, CDNA5 (gfx1250) wave32 + WMMA f32_16x16x32_f16.
// One workgroup (256 thr = 8 waves) per batch element; all intermediates live in LDS
// (~125 KB < 320 KB/WGP). Edge-MLP layer-1 concat-GEMM is algebraically split into two
// node-level GEMMs (Ai = h@W1[:64], Bj = h@W1[64:128]) + per-edge elementwise terms.
// Per-edge H x H GEMMs (m@W2, m@coord_W1) run on v_wmma_f32_16x16x32_f16 with fp32 acc.
#include <hip/hip_runtime.h>

typedef __attribute__((ext_vector_type(16))) _Float16 v16h;
typedef __attribute__((ext_vector_type(8)))  _Float16 v8h;
typedef __attribute__((ext_vector_type(8)))  float    v8f;

#define NN 64
#define HH 64
#define LL 4
#define BB 128

struct LayerPtrs {
  const float *edge_W1, *edge_b1, *edge_W2, *edge_b2;
  const float *att_W, *att_b;
  const float *coord_W1, *coord_b1, *coord_W2;
  const float *node_W1, *node_b1, *node_W2, *node_b2;
};
struct EgnnArgs {
  const float* x;
  const float* emb_W; const float* emb_b;
  LayerPtrs layers[LL];
  const float* coords_range; const float* output_scale;
  float* out;
};

struct Smem {
  float coord[NN][4];
  float x0[NN][4];
  float dcoord[NN][4];
  float h[NN][HH];                 // fp32 master copy of node features
  _Float16 hf16[NN][HH];
  _Float16 Ai[NN][HH];             // h @ edge_W1[0:64]
  _Float16 Bj[NN][HH];             // h @ edge_W1[64:128]
  _Float16 aggf16[NN][HH];         // sum_j gated messages
  _Float16 W1a[HH][HH];
  _Float16 W1b[HH][HH];
  _Float16 W2[HH][HH];
  _Float16 cW1[HH][HH];
  _Float16 nW1a[HH][HH];
  _Float16 nW1b[HH][HH];
  _Float16 nW2[HH][HH];
  union {
    _Float16 stage[8][16][HH];     // per-wave D->A transpose staging (edge phase)
    _Float16 updf16[NN][HH];       // node-MLP hidden (node phase)
  } u;
  float wr[HH], we[HH], b1[HH], b2[HH], attW[HH], cb1[HH], cW2v[HH], nb1[HH], nb2[HH];
  float attb, crange, oscale;
  float red[4];
};

__device__ __forceinline__ float siluf(float z) { return z / (1.0f + __expf(-z)); }
__device__ __forceinline__ float sigmf(float z) { return 1.0f / (1.0f + __expf(-z)); }

// 16-bit A-matrix (16x32) element->K map (ISA 7.12.2): lane<16 VGPRs 0..3 hold K=0..7,
// VGPRs 4..7 hold K=16..23; lanes>=16 are offset by +8. Elements come in two
// contiguous 8-half chunks, so fragments load as two 16-byte LDS reads.
__device__ __forceinline__ v16h load_a_frag(const _Float16* row, int kbase, int lane) {
  const int off = (lane & 16) ? 8 : 0;
  union { v16h v; v8h h[2]; } r;
  r.h[0] = *(const v8h*)(row + kbase + off);
  r.h[1] = *(const v8h*)(row + kbase + 16 + off);
  return r.v;
}
// 16-bit B-matrix (32x16): lane%16 = N, elements = K in the same order as the A fragment.
__device__ __forceinline__ v16h load_b_frag(_Float16 (*W)[HH], int kstep, int nt, int lane) {
  const int off = (lane & 16) ? 8 : 0;
  const int n = nt * 16 + (lane & 15);
  v16h r;
  #pragma unroll
  for (int e = 0; e < 16; ++e) r[e] = W[kstep * 32 + ((e < 8) ? e : e + 8) + off][n];
  return r;
}
__device__ __forceinline__ v8f wmma_f16(v16h a, v16h b, v8f c) {
  return __builtin_amdgcn_wmma_f32_16x16x32_f16(false, a, false, b, (short)0, c, false, false);
}

__global__ void __launch_bounds__(256, 1) egnn_fused(EgnnArgs args) {
  extern __shared__ char smem_raw[];
  Smem& s = *reinterpret_cast<Smem*>(smem_raw);
  const int t    = threadIdx.x;
  const int wave = t >> 5;
  const int lane = t & 31;
  const int hlf  = (lane >> 4) & 1;   // which 16-lane half
  const int ln   = lane & 15;
  const int b    = blockIdx.x;

  // ---------------- init: coords + h = emb(ones) ----------------
  if (t < NN * 3) {
    int n = t / 3, d = t % 3;
    float v = args.x[b * (NN * 3) + n * 3 + d];
    s.x0[n][d]    = v;
    s.coord[n][d] = v;
  }
  {
    float* hp = &s.h[0][0];
    for (int k = t; k < NN * HH; k += 256) {
      int f = k & (HH - 1);
      hp[k] = args.emb_W[f] + args.emb_b[f];      // all nodes identical at t=0
    }
  }
  if (t == 0) { s.crange = args.coords_range[0]; s.oscale = args.output_scale[0]; }
  __syncthreads();

  for (int layer = 0; layer < LL; ++layer) {
    const LayerPtrs lp = args.layers[layer];
    // -------- preamble: weights -> LDS (f16), h -> f16 --------
    {
      _Float16* w1a = &s.W1a[0][0];  _Float16* w1b = &s.W1b[0][0];
      _Float16* w2  = &s.W2[0][0];   _Float16* cw1 = &s.cW1[0][0];
      _Float16* n1a = &s.nW1a[0][0]; _Float16* n1b = &s.nW1b[0][0];
      _Float16* n2  = &s.nW2[0][0];
      for (int k = t; k < HH * HH; k += 256) {
        w1a[k] = (_Float16)lp.edge_W1[k];
        w1b[k] = (_Float16)lp.edge_W1[HH * HH + k];
        w2[k]  = (_Float16)lp.edge_W2[k];
        cw1[k] = (_Float16)lp.coord_W1[k];
        n1a[k] = (_Float16)lp.node_W1[k];
        n1b[k] = (_Float16)lp.node_W1[HH * HH + k];
        n2[k]  = (_Float16)lp.node_W2[k];
      }
      if (t < HH) {
        s.wr[t]   = lp.edge_W1[128 * HH + t];
        s.we[t]   = lp.edge_W1[129 * HH + t];
        s.b1[t]   = lp.edge_b1[t];
        s.b2[t]   = lp.edge_b2[t];
        s.attW[t] = lp.att_W[t];
        s.cb1[t]  = lp.coord_b1[t];
        s.cW2v[t] = lp.coord_W2[t];
        s.nb1[t]  = lp.node_b1[t];
        s.nb2[t]  = lp.node_b2[t];
      }
      if (t == 0) s.attb = lp.att_b[0];
      _Float16* hf = &s.hf16[0][0];
      const float* hp = &s.h[0][0];
      for (int k = t; k < NN * HH; k += 256) hf[k] = (_Float16)hp[k];
    }
    __syncthreads();

    // -------- node pre-GEMMs: Ai = h@W1a, Bj = h@W1b (32 16x16 tiles / 8 waves) --------
    for (int q = 0; q < 4; ++q) {
      int jb = wave * 4 + q;
      int mat = jb >> 4, mt = (jb >> 2) & 3, nt = jb & 3;
      _Float16 (*W)[HH] = mat ? s.W1b : s.W1a;
      v8f acc = {};
      #pragma unroll
      for (int ks = 0; ks < 2; ++ks)
        acc = wmma_f16(load_a_frag(&s.hf16[mt * 16 + ln][0], ks * 32, lane),
                       load_b_frag(W, ks, nt, lane), acc);
      _Float16 (*dst)[HH] = mat ? s.Bj : s.Ai;
      int row0 = mt * 16 + 8 * hlf;
      #pragma unroll
      for (int r = 0; r < 8; ++r) dst[row0 + r][nt * 16 + ln] = (_Float16)acc[r];
    }
    __syncthreads();

    // -------- edge phase: each wave owns 8 source nodes i --------
    for (int ii = 0; ii < 8; ++ii) {
      const int i = wave * 8 + ii;
      const float cix = s.coord[i][0], ciy = s.coord[i][1], ciz = s.coord[i][2];
      const float xix = s.x0[i][0],    xiy = s.x0[i][1],    xiz = s.x0[i][2];
      float aggacc[4] = {0, 0, 0, 0};
      float dacc[3]   = {0, 0, 0};
      // hoist Ai[i] + b1 in this lane's A-fragment feature set
      float aib0[16], aib1[16];
      #pragma unroll
      for (int e = 0; e < 16; ++e) {
        int v = e >> 1, p = e & 1;
        int f0 = ((v < 4) ? (2 * v) : (16 + 2 * (v - 4))) + 8 * hlf + p;
        aib0[e] = (float)s.Ai[i][f0]      + s.b1[f0];
        aib1[e] = (float)s.Ai[i][f0 + 32] + s.b1[f0 + 32];
      }
      for (int jt = 0; jt < 4; ++jt) {
        const int jA = jt * 16 + ln;   // this lane's edge row (A-layout M = lane%16)
        float dx = cix - s.coord[jA][0], dy = ciy - s.coord[jA][1], dz = ciz - s.coord[jA][2];
        float rad = dx * dx + dy * dy + dz * dz;
        float ex = xix - s.x0[jA][0], ey = xiy - s.x0[jA][1], ez = xiz - s.x0[jA][2];
        float ea = ex * ex + ey * ey + ez * ez;
        // build silu(edge-MLP layer 1) directly in WMMA A layout (no staging)
        v16h a0, a1;
        #pragma unroll
        for (int e = 0; e < 16; ++e) {
          int v = e >> 1, p = e & 1;
          int f0 = ((v < 4) ? (2 * v) : (16 + 2 * (v - 4))) + 8 * hlf + p;
          float z0 = aib0[e] + (float)s.Bj[jA][f0]      + rad * s.wr[f0]      + ea * s.we[f0];
          float z1 = aib1[e] + (float)s.Bj[jA][f0 + 32] + rad * s.wr[f0 + 32] + ea * s.we[f0 + 32];
          a0[e] = (_Float16)siluf(z0);
          a1[e] = (_Float16)siluf(z1);
        }
        // m2 = silu(m1 @ W2 + b2); attention partials on the fly
        v8f m2t[4];
        float attp[8] = {0, 0, 0, 0, 0, 0, 0, 0};
        #pragma unroll
        for (int nt = 0; nt < 4; ++nt) {
          float bb = s.b2[nt * 16 + ln];
          v8f acc = {bb, bb, bb, bb, bb, bb, bb, bb};
          acc = wmma_f16(a0, load_b_frag(s.W2, 0, nt, lane), acc);
          acc = wmma_f16(a1, load_b_frag(s.W2, 1, nt, lane), acc);
          float aw = s.attW[nt * 16 + ln];
          #pragma unroll
          for (int r = 0; r < 8; ++r) {
            float m2v = siluf(acc[r]);
            acc[r] = m2v;
            attp[r] += m2v * aw;
          }
          m2t[nt] = acc;
        }
        // reduce att over the 16 lanes of each half -> per-edge gate
        #pragma unroll
        for (int r = 0; r < 8; ++r) {
          float v = attp[r];
          v += __shfl_xor(v, 1, 32);
          v += __shfl_xor(v, 2, 32);
          v += __shfl_xor(v, 4, 32);
          v += __shfl_xor(v, 8, 32);
          attp[r] = sigmf(v + s.attb);
        }
        // gate, self-edge mask, agg accumulate, stage f16 row-major for transpose
        _Float16 (*st)[HH] = s.u.stage[wave];
        #pragma unroll
        for (int nt = 0; nt < 4; ++nt) {
          v8f g = m2t[nt];
          float ssum = 0.0f;
          #pragma unroll
          for (int r = 0; r < 8; ++r) {
            int e = r + 8 * hlf;                         // edge index within tile (D-layout M)
            float msk = ((jt * 16 + e) == i) ? 0.0f : 1.0f;
            float mv = g[r] * attp[r] * msk;
            ssum += mv;
            st[e][nt * 16 + ln] = (_Float16)mv;
          }
          ssum += __shfl_xor(ssum, 16, 32);
          aggacc[nt] += ssum;
        }
        asm volatile("s_wait_dscnt 0" ::: "memory");     // wave-local LDS RAW fence
        // coord MLP: c1 = silu(m_gated @ cW1 + cb1); cdot = c1 . cW2
        v16h ca0 = load_a_frag(&st[ln][0], 0, lane);
        v16h ca1 = load_a_frag(&st[ln][0], 32, lane);
        float cdot[8] = {0, 0, 0, 0, 0, 0, 0, 0};
        #pragma unroll
        for (int nt = 0; nt < 4; ++nt) {
          float bb = s.cb1[nt * 16 + ln];
          v8f acc = {bb, bb, bb, bb, bb, bb, bb, bb};
          acc = wmma_f16(ca0, load_b_frag(s.cW1, 0, nt, lane), acc);
          acc = wmma_f16(ca1, load_b_frag(s.cW1, 1, nt, lane), acc);
          float cw = s.cW2v[nt * 16 + ln];
          #pragma unroll
          for (int r = 0; r < 8; ++r) cdot[r] += siluf(acc[r]) * cw;
        }
        #pragma unroll
        for (int r = 0; r < 8; ++r) {
          float v = cdot[r];
          v += __shfl_xor(v, 1, 32);
          v += __shfl_xor(v, 2, 32);
          v += __shfl_xor(v, 4, 32);
          v += __shfl_xor(v, 8, 32);
          cdot[r] = tanhf(v) * s.crange;
        }
        if (ln == 0) {                                   // one lane per half accumulates
          #pragma unroll
          for (int r = 0; r < 8; ++r) {
            int j = jt * 16 + r + 8 * hlf;
            float c2 = cdot[r];
            dacc[0] += (cix - s.coord[j][0]) * c2;
            dacc[1] += (ciy - s.coord[j][1]) * c2;
            dacc[2] += (ciz - s.coord[j][2]) * c2;
          }
        }
      } // jt
      if (lane < 16) {
        #pragma unroll
        for (int nt = 0; nt < 4; ++nt)
          s.aggf16[i][nt * 16 + lane] = (_Float16)aggacc[nt];
      }
      #pragma unroll
      for (int d = 0; d < 3; ++d) {
        float v = dacc[d];
        v += __shfl_xor(v, 16, 32);                      // combine the two halves
        if (lane == 0) s.dcoord[i][d] = v;
      }
    } // ii
    __syncthreads();

    // -------- node MLP GEMM 1: upd = silu([h, agg] @ nW1 + nb1) --------
    for (int q = 0; q < 2; ++q) {
      int jb = wave * 2 + q;
      int mt = jb >> 2, nt = jb & 3;
      float bb = s.nb1[nt * 16 + ln];
      v8f acc = {bb, bb, bb, bb, bb, bb, bb, bb};
      #pragma unroll
      for (int ks = 0; ks < 2; ++ks) {
        acc = wmma_f16(load_a_frag(&s.hf16[mt * 16 + ln][0], ks * 32, lane),
                       load_b_frag(s.nW1a, ks, nt, lane), acc);
        acc = wmma_f16(load_a_frag(&s.aggf16[mt * 16 + ln][0], ks * 32, lane),
                       load_b_frag(s.nW1b, ks, nt, lane), acc);
      }
      int row0 = mt * 16 + 8 * hlf;
      #pragma unroll
      for (int r = 0; r < 8; ++r)
        s.u.updf16[row0 + r][nt * 16 + ln] = (_Float16)siluf(acc[r]);
    }
    __syncthreads();
    // coordinate update (dcoord complete; coord not read again until next-layer barrier)
    if (t < NN * 3) {
      int n = t / 3, d = t % 3;
      s.coord[n][d] += s.dcoord[n][d] * (1.0f / (NN - 1));
    }
    // -------- node MLP GEMM 2: h += upd @ nW2 + nb2 (residual) --------
    for (int q = 0; q < 2; ++q) {
      int jb = wave * 2 + q;
      int mt = jb >> 2, nt = jb & 3;
      float bb = s.nb2[nt * 16 + ln];
      v8f acc = {bb, bb, bb, bb, bb, bb, bb, bb};
      acc = wmma_f16(load_a_frag(&s.u.updf16[mt * 16 + ln][0], 0, lane),
                     load_b_frag(s.nW2, 0, nt, lane), acc);
      acc = wmma_f16(load_a_frag(&s.u.updf16[mt * 16 + ln][0], 32, lane),
                     load_b_frag(s.nW2, 1, nt, lane), acc);
      int row0 = mt * 16 + 8 * hlf;
      #pragma unroll
      for (int r = 0; r < 8; ++r)
        s.h[row0 + r][nt * 16 + ln] += acc[r];
    }
    __syncthreads();
  } // layers

  // -------- output: vel = coord - x0, remove COM, scale --------
  if (t < 3) {
    float sm = 0.0f;
    for (int n = 0; n < NN; ++n) sm += s.coord[n][t] - s.x0[n][t];
    s.red[t] = sm * (1.0f / NN);
  }
  __syncthreads();
  if (t < NN * 3) {
    int n = t / 3, d = t % 3;
    args.out[b * (NN * 3) + n * 3 + d] =
        (s.coord[n][d] - s.x0[n][d] - s.red[d]) * s.oscale;
  }
}

extern "C" void kernel_launch(void* const* d_in, const int* in_sizes, int n_in,
                              void* d_out, int out_size, void* d_ws, size_t ws_size,
                              hipStream_t stream) {
  (void)in_sizes; (void)n_in; (void)d_ws; (void)ws_size; (void)out_size;
  // Assumed flattening order (setup_inputs insertion order):
  // x, emb_W, emb_b, out_W, out_b,
  // [edge_W1, edge_b1, edge_W2, edge_b2, att_W, att_b, coord_W1, coord_b1,
  //  coord_W2, node_W1, node_b1, node_W2, node_b2] x 4,  coords_range, output_scale
  EgnnArgs a;
  a.x     = (const float*)d_in[0];
  a.emb_W = (const float*)d_in[1];
  a.emb_b = (const float*)d_in[2];
  int idx = 5;                                     // skip out_W, out_b (unused)
  for (int l = 0; l < LL; ++l) {
    a.layers[l].edge_W1  = (const float*)d_in[idx + 0];
    a.layers[l].edge_b1  = (const float*)d_in[idx + 1];
    a.layers[l].edge_W2  = (const float*)d_in[idx + 2];
    a.layers[l].edge_b2  = (const float*)d_in[idx + 3];
    a.layers[l].att_W    = (const float*)d_in[idx + 4];
    a.layers[l].att_b    = (const float*)d_in[idx + 5];
    a.layers[l].coord_W1 = (const float*)d_in[idx + 6];
    a.layers[l].coord_b1 = (const float*)d_in[idx + 7];
    a.layers[l].coord_W2 = (const float*)d_in[idx + 8];
    a.layers[l].node_W1  = (const float*)d_in[idx + 9];
    a.layers[l].node_b1  = (const float*)d_in[idx + 10];
    a.layers[l].node_W2  = (const float*)d_in[idx + 11];
    a.layers[l].node_b2  = (const float*)d_in[idx + 12];
    idx += 13;
  }
  a.coords_range = (const float*)d_in[idx + 0];
  a.output_scale = (const float*)d_in[idx + 1];
  a.out = (float*)d_out;
  egnn_fused<<<dim3(BB), dim3(256), sizeof(Smem), stream>>>(a);
}
